// QuantLinear_86878598463889
// MI455X (gfx1250) — compile-verified, matrix operand order checked by reading, not run on hardware
//
#include <hip/hip_runtime.h>

// ---- problem constants (match reference) ----
#define IN_F   8192
#define OUT_F  8192
#define TOK    16
#define KC     1024             // K-chunk staged in LDS (f16 elements)
#define LDSS   1032             // LDS row stride in halves = 2064B = 516 dwords
#define XH_STRIDE 8208          // f16 x-copy row stride (16-half slack for TDM over-read)

typedef _Float16 v16h __attribute__((ext_vector_type(16)));
typedef _Float16 v8h  __attribute__((ext_vector_type(8)));
typedef float    v8f  __attribute__((ext_vector_type(8)));
typedef float    v4f  __attribute__((ext_vector_type(4)));
typedef unsigned v4u  __attribute__((ext_vector_type(4)));
typedef int      v8i  __attribute__((ext_vector_type(8)));
typedef int      v4i  __attribute__((ext_vector_type(4)));

union AFrag { v16h v; v8h h2[2]; };
union BFrag { v16h v; unsigned u[8]; };

// ---------------------------------------------------------------------------
// Kernel 1: per-token prep. Converts x to f16 with the within-8 K-permutation
// {0,4,1,5,2,6,3,7} (matches the nibble-pair dequant order) into xh, and
// computes rowsum[t] = sum_k x[t,k] (folds zero-point + the f16 +1024 bias).
// ---------------------------------------------------------------------------
__global__ void __launch_bounds__(256) prep16_kernel(const float* __restrict__ x,
                                                     float* __restrict__ rs,
                                                     _Float16* __restrict__ xh) {
  __shared__ float red[256];
  const int row = blockIdx.x;
  float s = 0.f;
  for (int g = threadIdx.x; g < IN_F / 8; g += 256) {
    const float* xp = &x[row * IN_F + g * 8];
    const v4f a0 = *(const v4f*)(xp);
    const v4f a1 = *(const v4f*)(xp + 4);
    s += (a0[0] + a0[1]) + (a0[2] + a0[3]) + (a1[0] + a1[1]) + (a1[2] + a1[3]);
    v8h h;
    h[0] = (_Float16)a0[0]; h[1] = (_Float16)a1[0];
    h[2] = (_Float16)a0[1]; h[3] = (_Float16)a1[1];
    h[4] = (_Float16)a0[2]; h[5] = (_Float16)a1[2];
    h[6] = (_Float16)a0[3]; h[7] = (_Float16)a1[3];
    *(v8h*)&xh[row * XH_STRIDE + g * 8] = h;
  }
  if (threadIdx.x < XH_STRIDE - IN_F)                 // define the slack bytes
    xh[row * XH_STRIDE + IN_F + threadIdx.x] = (_Float16)0.f;
  red[threadIdx.x] = s;
  __syncthreads();
  for (int off = 128; off > 0; off >>= 1) {
    if ((int)threadIdx.x < off) red[threadIdx.x] += red[threadIdx.x + off];
    __syncthreads();
  }
  if (threadIdx.x == 0) rs[row] = red[0];
}

// ---------------------------------------------------------------------------
// Shared inner loop: 32 WMMAs over one staged K-chunk (per wave, 16 N-columns).
// B built as f16(1024+n) via (pair & 0x000F000F) | 0x64006400 — and_or only.
// ---------------------------------------------------------------------------
__device__ __forceinline__ void wmma_ktile(const _Float16* __restrict__ xl,
                                           const unsigned* __restrict__ qw,
                                           int kc, int col, int m15, int rg,
                                           v8f& acc) {
#pragma unroll 4
  for (int kl = 0; kl < KC; kl += 32) {
    AFrag A;
    const _Float16* rp = &xl[m15 * LDSS + kl + rg * 8];
    A.h2[0] = *(const v8h*)(rp);
    A.h2[1] = *(const v8h*)(rp + 16);

    const int rowBase = (kc + kl) >> 3;
    const unsigned qa = qw[(rowBase + rg)     * OUT_F + col];
    const unsigned qb = qw[(rowBase + 2 + rg) * OUT_F + col];
    int pr = rowBase + 16 + rg; if (pr > (IN_F / 8) - 1) pr = (IN_F / 8) - 1;
    __builtin_prefetch(&qw[pr * OUT_F + col], 0, 1);   // global_prefetch_b8

    BFrag B;
    B.u[0] = ( qa        & 0x000F000Fu) | 0x64006400u;
    B.u[1] = ((qa >> 4)  & 0x000F000Fu) | 0x64006400u;
    B.u[2] = ((qa >> 8)  & 0x000F000Fu) | 0x64006400u;
    B.u[3] = ((qa >> 12) & 0x000F000Fu) | 0x64006400u;
    B.u[4] = ( qb        & 0x000F000Fu) | 0x64006400u;
    B.u[5] = ((qb >> 4)  & 0x000F000Fu) | 0x64006400u;
    B.u[6] = ((qb >> 8)  & 0x000F000Fu) | 0x64006400u;
    B.u[7] = ((qb >> 12) & 0x000F000Fu) | 0x64006400u;

    acc = __builtin_amdgcn_wmma_f32_16x16x32_f16(
        false, A.v, false, B.v, (short)0, acc, false, false);
  }
}

__device__ __forceinline__ void epilogue(v8f acc, const float* scales,
                                         const float* zeros, const float* bias,
                                         const float* rsum, float* out,
                                         int col, int rg) {
  const float s  = scales[col];
  const float z  = zeros[col];
  const float bv = bias[col];
  const float zz = 1024.0f * s + z;   // removes +1024 nibble bias and zero-point
#pragma unroll
  for (int v = 0; v < 8; ++v) {
    const int m = v + 8 * rg;         // C/D layout: VGPR v -> M, lane -> N
    out[m * OUT_F + col] = acc[v] * s - zz * rsum[m] + bv;
  }
}

// ---------------------------------------------------------------------------
// Kernel 2a (TDM path): each chunk of the pre-converted f16 x is staged into
// LDS by the Tensor Data Mover: one 2D tile, 16 rows x 516 dwords from a
// 4104-dword-stride tensor. Tile width 516 dwords = 2064B makes TDM's packed
// LDS rows land exactly on the bank-staggered LDSS layout (pad = over-read).
// ---------------------------------------------------------------------------
__global__ void __launch_bounds__(256) qlin16_wmma_tdm_kernel(
    const unsigned* __restrict__ qw, const float* __restrict__ scales,
    const float* __restrict__ zeros, const float* __restrict__ bias,
    const float* __restrict__ rsum, const _Float16* __restrict__ xh,
    float* __restrict__ out) {
  __shared__ _Float16 xlds[TOK * LDSS];   // 33,024 B

  const int lane = threadIdx.x & 31;
  const int wave = threadIdx.x >> 5;
  const int m15  = lane & 15;
  const int rg   = lane >> 4;
  const int col  = blockIdx.x * 128 + wave * 16 + m15;

  const unsigned long long xh_base = (unsigned long long)(uintptr_t)xh;
  const unsigned           lds_base = (unsigned)(uintptr_t)&xlds[0];

  v8f acc = {};

  for (int kc = 0; kc < IN_F; kc += KC) {
    __syncthreads();                       // previous chunk fully consumed
    if (wave == 0) {
      const unsigned long long ga = xh_base + (unsigned long long)kc * 2ull;
      v4u g0;                              // D# group 0 (ISA 8.3)
      g0[0] = 1u;                                           // count=1, user mode
      g0[1] = lds_base;                                     // lds_addr
      g0[2] = (unsigned)(ga & 0xFFFFFFFFull);               // global_addr lo
      g0[3] = (unsigned)((ga >> 32) & 0x01FFFFFFull) | 0x80000000u; // hi | type=2
      v8i g1;                              // D# group 1 (ISA 8.4)
      g1[0] = 0x00020000;                  // wg_mask=0, data_size=2 (4B)
      g1[1] = (int)(516u << 16);           // tensor_dim0 = 516 dwords (bits 79:48)
      g1[2] = (int)(16u  << 16);           // tensor_dim1 = 16      (bits 111:80)
      g1[3] = (int)(516u << 16);           // tile_dim0  = 516      (bits 127:112)
      g1[4] = 16;                          // tile_dim1 = 16, tile_dim2 = 0
      g1[5] = 4104;                        // tensor_dim0_stride = 4104 dwords
      g1[6] = 0;                           // dim0_stride hi | dim1_stride lo
      g1[7] = 0;
      const v4i zz4 = {0, 0, 0, 0};        // groups 2/3 unused (2D tile)
      const v8i zz8 = {0, 0, 0, 0, 0, 0, 0, 0};
      __builtin_amdgcn_tensor_load_to_lds(g0, g1, zz4, zz4, zz8, 0); // 6-arg form
      __builtin_amdgcn_s_wait_tensorcnt(0);
    }
    __syncthreads();                       // publish chunk to all waves
    wmma_ktile(xlds, qw, kc, col, m15, rg, acc);
  }
  epilogue(acc, scales, zeros, bias, rsum, out, col, rg);
}

// ---------------------------------------------------------------------------
// Kernel 2b (fallback, proven): VALU staging of f32 x each chunk.
// ---------------------------------------------------------------------------
__global__ void __launch_bounds__(256) qlin16_wmma_kernel(
    const float* __restrict__ x, const unsigned* __restrict__ qw,
    const float* __restrict__ scales, const float* __restrict__ zeros,
    const float* __restrict__ bias, const float* __restrict__ rsum,
    float* __restrict__ out) {
  __shared__ _Float16 xlds[TOK * LDSS];

  const int lane = threadIdx.x & 31;
  const int wave = threadIdx.x >> 5;
  const int m15  = lane & 15;
  const int rg   = lane >> 4;
  const int col  = blockIdx.x * 128 + wave * 16 + m15;

  v8f acc = {};

  for (int kc = 0; kc < IN_F; kc += KC) {
    __syncthreads();
    for (int gi = threadIdx.x; gi < TOK * (KC / 8); gi += 256) {
      const int r  = gi >> 7;
      const int gc = (gi & 127) << 3;
      const float* xp = &x[r * IN_F + kc + gc];
      const v4f a0 = *(const v4f*)(xp);
      const v4f a1 = *(const v4f*)(xp + 4);
      v8h h;
      h[0] = (_Float16)a0[0]; h[1] = (_Float16)a1[0];
      h[2] = (_Float16)a0[1]; h[3] = (_Float16)a1[1];
      h[4] = (_Float16)a0[2]; h[5] = (_Float16)a1[2];
      h[6] = (_Float16)a0[3]; h[7] = (_Float16)a1[3];
      *(v8h*)&xlds[r * LDSS + gc] = h;
    }
    __syncthreads();
    wmma_ktile(xlds, qw, kc, col, m15, rg, acc);
  }
  epilogue(acc, scales, zeros, bias, rsum, out, col, rg);
}

extern "C" void kernel_launch(void* const* d_in, const int* in_sizes, int n_in,
                              void* d_out, int out_size, void* d_ws, size_t ws_size,
                              hipStream_t stream) {
  const float*    x      = (const float*)d_in[0];
  const unsigned* qw     = (const unsigned*)d_in[1];
  const float*    scales = (const float*)d_in[2];
  const float*    zeros  = (const float*)d_in[3];
  const float*    bias   = (const float*)d_in[4];
  float*          out    = (float*)d_out;

  float*     rsum = (float*)d_ws;                       // 16 floats
  _Float16*  xh   = (_Float16*)((char*)d_ws + 64);      // [16 x XH_STRIDE] f16
  const size_t need = 64 + (size_t)TOK * XH_STRIDE * sizeof(_Float16);

  if (ws_size >= need) {
    prep16_kernel<<<TOK, 256, 0, stream>>>(x, rsum, xh);
    qlin16_wmma_tdm_kernel<<<OUT_F / 128, 256, 0, stream>>>(
        qw, scales, zeros, bias, rsum, xh, out);
  } else {
    prep16_kernel<<<TOK, 256, 0, stream>>>(x, rsum, xh);
    // If ws can't hold xh, stage in-kernel instead (only rsum in ws).
    qlin16_wmma_kernel<<<OUT_F / 128, 256, 0, stream>>>(
        x, qw, scales, zeros, bias, rsum, out);
  }
}
